// AttentionBlock_64347200028945
// MI455X (gfx1250) — matrix operand order repaired and checked by reading, not compiled, not run on hardware
//
#include <hip/hip_runtime.h>
#include <hip/hip_bf16.h>
#include <math.h>

// ---------------------------------------------------------------------------
// AttentionBlock for MI455X (gfx1250):
//   K1: causal conv QKV as WMMA GEMM (A = x-patch, B = LDS-staged W^T),
//       rotary+scale folded into the q/k stores.
//   K2: windowed causal flash-attention; k & v context tiles staged by the
//       Tensor Data Mover (TDM) with LDS padding; P re-read with 16B LDS loads, k/v B operands per element.
//   K3: 1x1 projection (WMMA) + BN stats, input tile staged by TDM.
//   K4: BatchNorm finalize.
// ---------------------------------------------------------------------------

typedef __attribute__((ext_vector_type(16))) _Float16 v16h;
typedef __attribute__((ext_vector_type(8)))  _Float16 v8h;
typedef __attribute__((ext_vector_type(8)))  float    v8f;
typedef __attribute__((ext_vector_type(4)))  unsigned int u32x4;
typedef __attribute__((ext_vector_type(8)))  int      i32x8;
typedef __attribute__((ext_vector_type(4)))  int      i32x4;

union H16 { v16h v; v8h h[2]; };

#define BB 16
#define DD 128
#define NN 8192
#define WW 128

__device__ __forceinline__ v8f wmma_f16(v16h a, v16h b, v8f c) {
  return __builtin_amdgcn_wmma_f32_16x16x32_f16(false, a, false, b, (short)0, c,
                                                false, false);
}

__device__ __forceinline__ float rowmax16(float v) {
  for (int off = 1; off < 16; off <<= 1) v = fmaxf(v, __shfl_xor(v, off, 32));
  return v;
}
__device__ __forceinline__ float rowsum16(float v) {
  for (int off = 1; off < 16; off <<= 1) v += __shfl_xor(v, off, 32);
  return v;
}

// ---------------------------------------------------------------------------
// TDM 2D tile load: tile1 rows x tile0 f16 elems, global row stride stride0,
// into LDS with pad_amt DWORDs inserted every pad_int-coded DWORDs.
// D# per CDNA5 ISA ch.8 (group0: count/lds/global addr, group1: dims/pad).
// ---------------------------------------------------------------------------
__device__ __forceinline__ void tdm_load_2d(const void* gsrc, unsigned lds_off,
                                            unsigned tile0, unsigned tile1,
                                            unsigned stride0, unsigned tdim0,
                                            unsigned pad_int, unsigned pad_amt) {
  unsigned long long ga = (unsigned long long)(size_t)gsrc;
  u32x4 g0;
  g0[0] = 1u;                                   // count=1 (user descriptor)
  g0[1] = lds_off;                              // lds_addr (bytes)
  g0[2] = (unsigned)ga;                         // global_addr[31:0]
  g0[3] = (unsigned)((ga >> 32) & 0x01ffffffu) | 0x80000000u;  // addr|type=2
  i32x8 g1;
  g1[0] = (int)((1u << 16)            // data_size = 2 bytes
              | (1u << 20)            // pad_enable
              | (pad_int << 22)       // pad interval code
              | (pad_amt << 25));     // pad amount code
  g1[1] = (int)(tdim0 << 16);                       // tensor_dim0 lo
  g1[2] = (int)((tdim0 >> 16) | (0x4000u << 16));   // tensor_dim1 = 16384
  g1[3] = (int)(tile0 << 16);                       // tile_dim0
  g1[4] = (int)tile1;                               // tile_dim1
  g1[5] = (int)stride0;                             // tensor_dim0_stride
  g1[6] = 0;
  g1[7] = 0;
  i32x4 z = {0, 0, 0, 0};
#if __clang_major__ >= 23
  i32x8 z8 = {0, 0, 0, 0, 0, 0, 0, 0};
  __builtin_amdgcn_tensor_load_to_lds(g0, g1, z, z, z8, 0);
#else
  __builtin_amdgcn_tensor_load_to_lds(g0, g1, z, z, 0);
#endif
}

// ---------------------------------------------------------------------------
// Kernel 1: fused causal conv1d for q,k,v with rotary folded into q/k stores.
// One WG per (batch, 128-position tile). Wave w owns positions [16w,16w+16),
// all 128 output channels (8 accumulator tiles).
// A-frag = x-patch rows (tap shift moves the row -> always 16B aligned);
// B-frag = W^T slice staged in LDS (64-channel halves).
// ---------------------------------------------------------------------------
__global__ __launch_bounds__(256) void qkv_conv_kernel(
    const float* __restrict__ x,
    const float* __restrict__ qw, const float* __restrict__ qb,
    const float* __restrict__ kw, const float* __restrict__ kb,
    const float* __restrict__ vw, const float* __restrict__ vb,
    _Float16* __restrict__ qh, _Float16* __restrict__ khT,
    _Float16* __restrict__ vh) {
  __shared__ _Float16 xp[132 * 136];  // [patch pos][channel], 272B pitch
  __shared__ _Float16 wt[64 * 128];   // [c within half][out ch], 256B pitch

  const int b = blockIdx.x >> 6;
  const int nt = blockIdx.x & 63;
  const int nbase = nt * 128;
  const int tid = threadIdx.x, lane = tid & 31, wave = tid >> 5;
  const int seg = (lane & 16) >> 1;  // 0 / 8: K- and M-offset for upper lanes
  const float scale = 0.08838834764831845f;  // 128^-0.5
  const float lf = 0.14391156509695413f;     // ln(10000)/64

  // stage x patch, position-major (row p -> pos nbase-4+p)
  for (int idx = tid; idx < 128 * 132; idx += 256) {
    int c = idx / 132, p = idx - c * 132;
    int pos = nbase - 4 + p;
    float v = (pos >= 0) ? x[((size_t)b * DD + c) * NN + pos] : 0.0f;
    xp[p * 136 + c] = (_Float16)v;
  }

  const float* wptr[3] = {qw, kw, vw};
  const float* bptr[3] = {qb, kb, vb};

  float invf[8];
  for (int ct = 0; ct < 8; ++ct)
    invf[ct] = __expf(-(float)((ct * 16 + (lane & 15)) & 63) * lf);

  const int posbase = nbase + wave * 16 + seg;

  for (int m = 0; m < 3; ++m) {
    const float* wm = wptr[m];
    if (m < 2) __builtin_prefetch(wptr[m + 1] + tid * 64, 0, 1);
    float biasv[8];
    for (int ct = 0; ct < 8; ++ct) biasv[ct] = bptr[m][ct * 16 + (lane & 15)];
    v8f acc[8];
    v8f z = {};
    for (int i = 0; i < 8; ++i) acc[i] = z;

    for (int t = 0; t < 5; ++t)
      for (int chh = 0; chh < 2; ++chh) {
        __syncthreads();
        // stage W^T half-slice: wt[c2][o] = w[o][chh*64+c2][t]
        for (int idx = tid; idx < 64 * 128; idx += 256) {
          int c2 = idx >> 7, o = idx & 127;
          wt[c2 * 128 + o] =
              (_Float16)wm[((size_t)o * DD + chh * 64 + c2) * 5 + t];
        }
        __syncthreads();
        for (int kk = 0; kk < 2; ++kk) {
          H16 af;  // A: M = position, K = channel chunk
          int p = wave * 16 + (lane & 15) + t;
          const _Float16* ar = &xp[p * 136 + (chh * 2 + kk) * 32];
          af.h[0] = *(const v8h*)(ar + seg);
          af.h[1] = *(const v8h*)(ar + 16 + seg);
          for (int ct = 0; ct < 8; ++ct) {
            H16 bf;  // B: lane%16 = out-channel, elements = K (channel)
            const _Float16* br = &wt[(kk * 32 + seg) * 128 + ct * 16 + (lane & 15)];
            for (int s = 0; s < 16; ++s)
              bf.v[s] = br[((s & 7) + ((s >> 3) << 4)) * 128];
            acc[ct] = wmma_f16(af.v, bf.v, acc[ct]);
          }
        }
      }

    // C layout: reg r -> position row, lane%16 -> out-channel col
    if (m == 0) {  // q: bias + rotary + scale -> (B,N,D)
      for (int r = 0; r < 8; ++r) {
        int pos = posbase + r;
        _Float16* qrow = qh + ((size_t)b * NN + pos) * DD + (lane & 15);
        for (int ct = 0; ct < 8; ++ct) {
          float a0 = acc[ct][r] + biasv[ct];
          float ap = acc[ct ^ 4][r] + biasv[ct ^ 4];
          float rot = (ct < 4) ? -ap : ap;
          float f = (float)pos * invf[ct];
          qrow[ct * 16] = (_Float16)((a0 * __cosf(f) + rot * __sinf(f)) * scale);
        }
      }
    } else if (m == 1) {  // k: bias + rotary -> transposed (B,D,N), 16B stores
      for (int ct = 0; ct < 8; ++ct) {
        v8h kv;
        for (int r = 0; r < 8; ++r) {
          int pos = posbase + r;
          float a0 = acc[ct][r] + biasv[ct];
          float ap = acc[ct ^ 4][r] + biasv[ct ^ 4];
          float rot = (ct < 4) ? -ap : ap;
          float f = (float)pos * invf[ct];
          kv[r] = (_Float16)(a0 * __cosf(f) + rot * __sinf(f));
        }
        int ch = ct * 16 + (lane & 15);
        *(v8h*)(khT + ((size_t)b * DD + ch) * NN + posbase) = kv;
      }
    } else {  // v: bias only -> (B,N,D)
      for (int r = 0; r < 8; ++r) {
        int pos = posbase + r;
        _Float16* vrow = vh + ((size_t)b * NN + pos) * DD + (lane & 15);
        for (int ct = 0; ct < 8; ++ct)
          vrow[ct * 16] = (_Float16)(acc[ct][r] + biasv[ct]);
      }
    }
  }
}

// ---------------------------------------------------------------------------
// Kernel 2: local causal attention, flash-style online softmax.
// One WG per (batch, window). Wave w owns 16 query rows. Both context tiles
// (k transposed, v row-major) are staged by the Tensor Data Mover.
// ---------------------------------------------------------------------------
__global__ __launch_bounds__(256) void local_attn_kernel(
    const _Float16* __restrict__ qh, const _Float16* __restrict__ khT,
    const _Float16* __restrict__ vh, _Float16* __restrict__ ahT) {
  __shared__ _Float16 kt[128 * 80];       // [d][j], TDM pad 8/32 -> 160B pitch
  __shared__ _Float16 vs[64 * 136];       // [j][d], TDM pad 4/64 -> 272B pitch
  __shared__ _Float16 pscr[8 * 16 * 32];  // per-wave P transpose scratch

  const int b = blockIdx.x >> 6;
  const int wi = blockIdx.x & 63;
  const int qbase = wi * WW;
  const int tid = threadIdx.x, lane = tid & 31, wave = tid >> 5;
  const int qrow0 = qbase + wave * 16;
  const int seg = (lane & 16) >> 1;

  // q A-fragments: rotary/scale already applied by kernel 1 -> pure 16B loads
  H16 QA[4];
  {
    const _Float16* qrow = qh + ((size_t)b * NN + qrow0 + (lane & 15)) * DD;
    for (int kc = 0; kc < 4; ++kc) {
      QA[kc].h[0] = *(const v8h*)(qrow + kc * 32 + seg);
      QA[kc].h[1] = *(const v8h*)(qrow + kc * 32 + 16 + seg);
    }
  }

  v8f acc[8];
  v8f zero = {};
  for (int i = 0; i < 8; ++i) acc[i] = zero;
  float mrow[8], lrow[8];
  for (int r = 0; r < 8; ++r) { mrow[r] = -1e30f; lrow[r] = 0.0f; }

  for (int jc = 0; jc < 4; ++jc) {
    __syncthreads();
    const int jpc = qbase - WW + jc * 64;  // first context row of this chunk
    // Rows < 0 are clamped to 0: their attention weights are exactly zero
    // after masking, so the (valid, deterministic) values are never used.
    if (wave == 0) {
      int gs = (jpc < 0) ? 0 : jpc;
      tdm_load_2d(khT + (size_t)b * DD * NN + gs, (unsigned)(size_t)kt,
                  /*tile0=*/64, /*tile1=*/128, /*stride0=*/NN, /*tdim0=*/NN,
                  /*pad_int=*/4, /*pad_amt=*/7);  // 32+8 DW -> 80-half pitch
      tdm_load_2d(vh + ((size_t)b * NN + gs) * DD, (unsigned)(size_t)vs,
                  /*tile0=*/128, /*tile1=*/64, /*stride0=*/DD, /*tdim0=*/DD,
                  /*pad_int=*/5, /*pad_amt=*/3);  // 64+4 DW -> 136-half pitch
      __builtin_amdgcn_s_wait_tensorcnt(0);
    }
    __syncthreads();

    for (int hc = 0; hc < 2; ++hc) {
      int jb = hc * 32;
      // sim = q . k^T for a 16x32 block of context columns
      v8f S0 = {}, S1 = {};
      for (int kc = 0; kc < 4; ++kc) {
        H16 b0, b1;
        const _Float16* kr = &kt[(kc * 32 + seg) * 80 + jb + (lane & 15)];  // elements = K (head dim)
        for (int s = 0; s < 16; ++s) {
          int K = (s & 7) + ((s >> 3) << 4);
          b0.v[s] = kr[K * 80];
          b1.v[s] = kr[K * 80 + 16];
        }
        S0 = wmma_f16(QA[kc].v, b0.v, S0);
        S1 = wmma_f16(QA[kc].v, b1.v, S1);
      }
      // mask + online softmax update
      int col = lane & 15;
      int jpos0 = jpc + jb + col;
      int jpos1 = jpos0 + 16;
      float mnew[8];
      for (int r = 0; r < 8; ++r) {
        int qpos = qrow0 + r + seg;
        float s0 = (jpos0 >= 0 && jpos0 <= qpos) ? S0[r] : -1e30f;
        float s1 = (jpos1 >= 0 && jpos1 <= qpos) ? S1[r] : -1e30f;
        S0[r] = s0;
        S1[r] = s1;
        mnew[r] = rowmax16(fmaxf(fmaxf(s0, s1), mrow[r]));
      }
      _Float16* pw = pscr + wave * (16 * 32);
      for (int r = 0; r < 8; ++r) {
        float rs = __expf(mrow[r] - mnew[r]);  // ->0 leaving all-masked state
        float p0 = __expf(S0[r] - mnew[r]);
        float p1 = __expf(S1[r] - mnew[r]);
        lrow[r] = lrow[r] * rs + rowsum16(p0 + p1);
        mrow[r] = mnew[r];
        for (int dt = 0; dt < 8; ++dt) acc[dt][r] *= rs;
        int M = r + seg;
        pw[M * 32 + col] = (_Float16)p0;
        pw[M * 32 + col + 16] = (_Float16)p1;
      }
      asm volatile("s_wait_dscnt 0" ::: "memory");
      // re-read P as an A fragment: two aligned 16B segments per lane
      H16 pa;
      {
        int M = lane & 15;
        pa.h[0] = *(const v8h*)&pw[M * 32 + seg];
        pa.h[1] = *(const v8h*)&pw[M * 32 + 16 + seg];
      }
      // out += P . V
      for (int dt = 0; dt < 8; ++dt) {
        H16 bv;
        const _Float16* vr = &vs[(jb + seg) * 136 + dt * 16 + (lane & 15)];  // elements = K
        for (int s = 0; s < 16; ++s)
          bv.v[s] = vr[((s & 7) + ((s >> 3) << 4)) * 136];
        acc[dt] = wmma_f16(pa.v, bv.v, acc[dt]);
      }
    }
  }

  // normalize and write transposed (B,D,N) with packed 16B stores
  float inv[8];
  for (int r = 0; r < 8; ++r) inv[r] = 1.0f / lrow[r];
  for (int dt = 0; dt < 8; ++dt) {
    v8h ov;
    for (int r = 0; r < 8; ++r) ov[r] = (_Float16)(acc[dt][r] * inv[r]);
    int d = dt * 16 + (lane & 15);
    *(v8h*)(ahT + ((size_t)b * DD + d) * NN + qrow0 + seg) = ov;
  }
}

// ---------------------------------------------------------------------------
// Kernel 3: 1x1 projection (WMMA GEMM) + per-channel sum/sumsq for BN.
// Channel-major input tile staged by TDM (pad 4/64 -> 272B pitch).
// ---------------------------------------------------------------------------
__global__ __launch_bounds__(256) void proj_stats_kernel(
    const _Float16* __restrict__ ahT, const float* __restrict__ ow,
    const float* __restrict__ ob, float* __restrict__ y,
    float* __restrict__ stats) {
  __shared__ _Float16 as_[128 * 136];  // [channel][position], 272B pitch

  const int b = blockIdx.x >> 6;
  const int nt = blockIdx.x & 63;
  const int nbase = nt * 128;
  const int tid = threadIdx.x, lane = tid & 31, wave = tid >> 5;
  const int obase = wave * 16;
  const int seg = (lane & 16) >> 1;

  if (wave == 0) {
    tdm_load_2d(ahT + (size_t)b * DD * NN + nbase, (unsigned)(size_t)as_,
                /*tile0=*/128, /*tile1=*/128, /*stride0=*/NN, /*tdim0=*/NN,
                /*pad_int=*/5, /*pad_amt=*/3);
    __builtin_amdgcn_s_wait_tensorcnt(0);
  }
  __syncthreads();

  v16h A[4];
  {
    int M = lane & 15;
    for (int kc = 0; kc < 4; ++kc) {
      v16h a;
      for (int h = 0; h < 16; ++h) {
        int K = (h & 7) + ((h >> 3) << 4) + seg;
        a[h] = (_Float16)ow[(size_t)(obase + M) * DD + kc * 32 + K];
      }
      A[kc] = a;
    }
  }
  float biasr[8];
  for (int r = 0; r < 8; ++r) biasr[r] = ob[obase + r + seg];

  for (int ns = 0; ns < 8; ++ns) {
    v8f acc = {};
    for (int kc = 0; kc < 4; ++kc) {
      H16 bf;
      const _Float16* ar = &as_[(kc * 32 + seg) * 136 + ns * 16 + (lane & 15)];
      for (int s = 0; s < 16; ++s)
        bf.v[s] = ar[((s & 7) + ((s >> 3) << 4)) * 136];
      acc = wmma_f16(A[kc], bf.v, acc);
    }
    int pos = nbase + ns * 16 + (lane & 15);
    for (int r = 0; r < 8; ++r) {
      float yv = acc[r] + biasr[r];
      int ch = obase + r + seg;
      y[((size_t)b * DD + ch) * NN + pos] = yv;
      float s1 = rowsum16(yv);
      float s2 = rowsum16(yv * yv);
      if ((lane & 15) == 0) {
        atomicAdd(&stats[ch], s1);
        atomicAdd(&stats[128 + ch], s2);
      }
    }
  }
}

__global__ void zero_stats_kernel(float* stats) { stats[threadIdx.x] = 0.0f; }

// ---------------------------------------------------------------------------
// Kernel 4: BatchNorm finalize, in place on d_out. float4 accesses.
// ---------------------------------------------------------------------------
__global__ __launch_bounds__(256) void bn_kernel(float* __restrict__ y,
                                                 const float* __restrict__ stats,
                                                 const float* __restrict__ gamma,
                                                 const float* __restrict__ beta) {
  const float invc = 1.0f / (16.0f * 8192.0f);
  size_t idx = ((size_t)blockIdx.x * 256 + threadIdx.x) * 4;
  int ch = (int)((idx >> 13) & 127);
  float mean = stats[ch] * invc;
  float var = stats[128 + ch] * invc - mean * mean;
  float sc = rsqrtf(var + 1e-5f) * gamma[ch];
  float bt = beta[ch] - mean * sc;
  float4 v = *(float4*)(y + idx);
  v.x = v.x * sc + bt;
  v.y = v.y * sc + bt;
  v.z = v.z * sc + bt;
  v.w = v.w * sc + bt;
  *(float4*)(y + idx) = v;
}

// ---------------------------------------------------------------------------
extern "C" void kernel_launch(void* const* d_in, const int* in_sizes, int n_in,
                              void* d_out, int out_size, void* d_ws,
                              size_t ws_size, hipStream_t stream) {
  (void)in_sizes; (void)n_in; (void)out_size; (void)ws_size;
  const float* x  = (const float*)d_in[0];
  const float* qw = (const float*)d_in[1];
  const float* qb = (const float*)d_in[2];
  const float* kw = (const float*)d_in[3];
  const float* kb = (const float*)d_in[4];
  const float* vw = (const float*)d_in[5];
  const float* vb = (const float*)d_in[6];
  const float* ow = (const float*)d_in[7];
  const float* ob = (const float*)d_in[8];
  const float* gamma = (const float*)d_in[9];
  const float* beta  = (const float*)d_in[10];

  char* ws = (char*)d_ws;
  float* stats = (float*)ws;                                  // 256 floats
  const size_t HSZ = (size_t)BB * NN * DD * sizeof(_Float16); // 32 MB each
  _Float16* qh  = (_Float16*)(ws + 4096);
  _Float16* khT = (_Float16*)(ws + 4096 + HSZ);
  _Float16* vh  = (_Float16*)(ws + 4096 + 2 * HSZ);
  _Float16* ahT = (_Float16*)(ws + 4096 + 3 * HSZ);
  float* y = (float*)d_out;

  zero_stats_kernel<<<1, 256, 0, stream>>>(stats);
  qkv_conv_kernel<<<BB * (NN / 128), 256, 0, stream>>>(x, qw, qb, kw, kb, vw,
                                                       vb, qh, khT, vh);
  local_attn_kernel<<<BB * (NN / WW), 256, 0, stream>>>(qh, khT, vh, ahT);
  proj_stats_kernel<<<BB * (NN / 128), 256, 0, stream>>>(ahT, ow, ob, y, stats);
  bn_kernel<<<(BB * DD * NN) / (256 * 4), 256, 0, stream>>>(y, stats, gamma,
                                                            beta);
}
